// RNN_44074954392190
// MI455X (gfx1250) — matrix operand order repaired and checked by reading, not compiled, not run on hardware
//
#include <hip/hip_runtime.h>
#include <stdint.h>

namespace {
constexpr int   kB    = 64;
constexpr int   kT    = 1000;
constexpr int   kNin  = 16;
constexpr int   kNrec = 512;
constexpr int   kNout = 8;
constexpr int   kNwg  = 8;              // persistent workgroups
constexpr int   kCols = kNrec / kNwg;   // 64 output neurons per WG
constexpr int   kKch  = kNrec / 32;     // 16 K-chunks of 32
constexpr float kAlpha = 0.1f;
constexpr float kBeta  = 0.9f;
}

typedef __attribute__((ext_vector_type(16))) __bf16 v16bf;
typedef __attribute__((ext_vector_type(8)))  float  v8f;

union FragU {
  uint4          q[2];
  unsigned short s[16];
  v16bf          v;
};

__device__ __forceinline__ unsigned short f2bf(float f) {
  union { float f; unsigned u; } x;
  x.f = f;
  unsigned r = x.u + 0x7FFFu + ((x.u >> 16) & 1u);  // round-to-nearest-even
  return (unsigned short)(r >> 16);
}

// A-matrix fragment (16x32 bf16): lane holds row m=lane&15.
// elems 0..15 = row[k0 + koff .. +15] contiguous,
// koff = 16 for lanes 16..31 (matches B fragment K order).
__device__ __forceinline__ v16bf loadA_lane(const unsigned short* row, int k0, int hi) {
  FragU f;
  const int koff = hi ? 16 : 0;
  f.q[0] = *(const uint4*)(row + k0 + koff);
  f.q[1] = *(const uint4*)(row + k0 + koff + 8);
  return f.v;
}

// B-matrix fragment (32x16 bf16): lane holds col n=lane&15,
// K rows k0+16*hi .. +15 contiguous (row-major w[n][k]).
__device__ __forceinline__ v16bf loadB_lane(const unsigned short* row, int k0, int hi) {
  FragU f;
  const int kb = k0 + (hi ? 16 : 0);
  f.q[0] = *(const uint4*)(row + kb);
  f.q[1] = *(const uint4*)(row + kb + 8);
  return f.v;
}

__device__ __forceinline__ v16bf frag_of(uint4 a, uint4 b) {
  FragU f; f.q[0] = a; f.q[1] = b; return f.v;
}

__device__ __forceinline__ v8f wmma_bf16(v16bf a, v16bf b, v8f c) {
  return __builtin_amdgcn_wmma_f32_16x16x32_bf16(false, a, false, b, (short)0, c,
                                                 false, false);
}

// -------------------- helper kernels --------------------

__global__ void k_zero_f32(float* __restrict__ p, size_t n) {
  size_t i = (size_t)blockIdx.x * blockDim.x + threadIdx.x;
  const size_t st = (size_t)gridDim.x * blockDim.x;
  for (; i < n; i += st) p[i] = 0.f;
}

__global__ void k_zero_u32(unsigned* __restrict__ p, size_t n) {
  size_t i = (size_t)blockIdx.x * blockDim.x + threadIdx.x;
  const size_t st = (size_t)gridDim.x * blockDim.x;
  for (; i < n; i += st) p[i] = 0u;
}

// t=0 slices of hidden_act / a_list are h0 = e0 = 0
__global__ void k_init_t0(float* __restrict__ hid_act, float* __restrict__ alist) {
  const size_t base = (size_t)blockIdx.x * (kT + 1) * kNrec;
  for (int k = threadIdx.x; k < kNrec; k += blockDim.x) {
    hid_act[base + k] = 0.f;
    alist[base + k] = 0.f;
  }
}

// w_out -> bf16, padded to 16 rows (rows 8..15 are zero) so the readout
// B-fragment load needs no lane guard.
__global__ void k_cvt_wout(const float* __restrict__ w, unsigned short* __restrict__ o) {
  const int i = blockIdx.x * blockDim.x + threadIdx.x;
  if (i < 16 * kNrec) {
    const int row = i / kNrec;
    o[i] = (row < kNout) ? f2bf(w[i]) : (unsigned short)0;
  }
}

// -------------------- main persistent rollout --------------------
//
// grid = 8 WGs x 256 threads (8 waves). WG g owns output neurons [g*64, g*64+64).
// wave w: m-tile = w>>1 (batch rows), two n-tiles = (w&1)*2 + {0,1}.
// w_rec fragments stay resident in VGPRs for all 1000 steps.
// Per step: readout WMMA -> error forcing -> 16 recurrent WMMAs + 1 fused
// [u|err|1]x[w_in|w_fb|b_rec] WMMA -> leaky integrate, tanh, publish r (bf16)
// through a ping-pong L2 buffer, device-wide atomic barrier.

__global__ __launch_bounds__(256, 2) void ctrnn_rollout(
    const float* __restrict__ input, const float* __restrict__ y,
    const float* __restrict__ mask, const float* __restrict__ w_in,
    const float* __restrict__ w_rec, const float* __restrict__ w_fb,
    const float* __restrict__ b_rec, const float* __restrict__ b_out,
    float* __restrict__ hid_act, float* __restrict__ outputs,
    float* __restrict__ alist, unsigned* __restrict__ barrier_cnt,
    unsigned short* __restrict__ rping, const unsigned short* __restrict__ woutbf) {
  __shared__ unsigned short combA[kB][32];  // [u_t | err_t | 1 | pad] per batch row

  const int tid   = threadIdx.x;
  const int lane  = tid & 31;
  const int wv    = tid >> 5;        // 0..7
  const int wg    = blockIdx.x;      // 0..7
  const int mtile = wv >> 1;         // 0..3
  const int ntb   = (wv & 1) * 2;    // 0 or 2
  const int nl    = lane & 15;
  const int hi    = (lane >> 4) & 1;

  // ---- resident weight fragments: w_rec slice (2 n-tiles x 16 chunks) ----
  uint4 wrq[2][kKch][2];
  uint4 cwq[2][2];
#pragma unroll
  for (int i = 0; i < 2; ++i) {
    const int n = wg * kCols + (ntb + i) * 16 + nl;  // global output neuron
#pragma unroll
    for (int c = 0; c < kKch; ++c) {
      FragU f;
      const int kb = c * 32 + hi * 16;
#pragma unroll
      for (int e = 0; e < 16; ++e) f.s[e] = f2bf(w_rec[(size_t)n * kNrec + kb + e]);
      wrq[i][c][0] = f.q[0];
      wrq[i][c][1] = f.q[1];
    }
    // fused input/feedback/bias B fragment: K = [w_in(16) | w_fb(8) | b_rec(1) | 0]
    FragU f;
#pragma unroll
    for (int e = 0; e < 16; ++e) {
      const int kk = hi * 16 + e;
      float v = 0.f;
      if (kk < kNin)                 v = w_in[(size_t)n * kNin + kk];
      else if (kk < kNin + kNout)    v = w_fb[(size_t)n * kNout + (kk - kNin)];
      else if (kk == kNin + kNout)   v = b_rec[n];
      f.s[e] = f2bf(v);
    }
    cwq[i][0] = f.q[0];
    cwq[i][1] = f.q[1];
  }
  const float bo = (nl < kNout) ? b_out[nl] : 0.f;

  // static part of combA: k==24 is the bias lane (1.0), 25..31 pad with zero
  for (int idx = tid; idx < kB * 32; idx += 256) {
    const int b = idx >> 5, k = idx & 31;
    if (k >= 16) combA[b][k] = (k == 24) ? f2bf(1.0f) : (unsigned short)0;
  }

  // ---- recurrent state in accumulator layout ----
  v8f h[2], ee[2];
  float rl[2][8];
#pragma unroll
  for (int i = 0; i < 2; ++i)
#pragma unroll
    for (int j = 0; j < 8; ++j) { h[i][j] = 0.f; ee[i][j] = 0.f; rl[i][j] = 0.f; }

  const v8f vzero = {0.f, 0.f, 0.f, 0.f, 0.f, 0.f, 0.f, 0.f};

  __syncthreads();

#pragma unroll 1
  for (int t = 0; t < kT; ++t) {
    const unsigned short* rb_rd = rping + (size_t)(t & 1) * (kB * kNrec);
    unsigned short* rb_wr = rping + (size_t)((t + 1) & 1) * (kB * kNrec);

    // stage u_t (bf16) into combA[:, 0:16]  (1024 elems / 256 threads)
#pragma unroll
    for (int rr = 0; rr < 4; ++rr) {
      const int idx = tid * 4 + rr;
      const int b = idx >> 4, k = idx & 15;
      combA[b][k] = f2bf(input[((size_t)b * kT + t) * kNin + k]);
    }
    __syncthreads();

    // ---- readout + error forcing (waves 0..3; out-tile rows m = wv*16..) ----
    if (wv < 4) {
      const unsigned short* arow = rb_rd + (size_t)(wv * 16 + nl) * kNrec;
      const unsigned short* brow = woutbf + (size_t)nl * kNrec;  // zero-padded rows
      v8f oacc = vzero;
#pragma unroll
      for (int c = 0; c < kKch; ++c) {
        const v16bf a = loadA_lane(arow, c * 32, hi);
        const v16bf bm = loadB_lane(brow, c * 32, hi);
        oacc = wmma_bf16(a, bm, oacc);
      }
      if (nl < kNout) {
#pragma unroll
        for (int j = 0; j < 8; ++j) {
          const int b   = wv * 16 + j + hi * 8;
          const float o = oacc[j] + bo;
          const float yv = y[((size_t)b * kT + t) * kNout + nl];
          const float mk = mask[(size_t)b * kT + t];
          combA[b][kNin + nl] = f2bf((yv - o) * mk);  // err feeds next WMMA
          if (wg == 0) outputs[((size_t)b * kT + t) * kNout + nl] = o;
        }
      }
    }
    __syncthreads();

    // ---- recurrent gemm (resident B) + fused input/feedback/bias WMMA ----
    const unsigned short* arow = rb_rd + (size_t)(mtile * 16 + nl) * kNrec;
    v8f acc0 = vzero, acc1 = vzero;
#pragma unroll
    for (int c = 0; c < kKch; ++c) {
      const v16bf a = loadA_lane(arow, c * 32, hi);
      acc0 = wmma_bf16(a, frag_of(wrq[0][c][0], wrq[0][c][1]), acc0);
      acc1 = wmma_bf16(a, frag_of(wrq[1][c][0], wrq[1][c][1]), acc1);
    }
    {
      const v16bf a = loadA_lane(&combA[mtile * 16 + nl][0], 0, hi);
      acc0 = wmma_bf16(a, frag_of(cwq[0][0], cwq[0][1]), acc0);
      acc1 = wmma_bf16(a, frag_of(cwq[1][0], cwq[1][1]), acc1);
    }

    // ---- leaky integration, eligibility trace, publish r_{t+1} ----
#pragma unroll
    for (int i = 0; i < 2; ++i) {
      const v8f acc = i ? acc1 : acc0;
      const int n = wg * kCols + (ntb + i) * 16 + nl;
#pragma unroll
      for (int j = 0; j < 8; ++j) {
        const int b = mtile * 16 + j + hi * 8;
        const float hn = kBeta * h[i][j] + kAlpha * acc[j];
        const float en = kBeta * ee[i][j] + kAlpha * rl[i][j];  // uses r_t
        const float rn = tanhf(hn);
        h[i][j] = hn; ee[i][j] = en; rl[i][j] = rn;
        const size_t row = ((size_t)b * (kT + 1) + (t + 1)) * kNrec + n;
        __builtin_nontemporal_store(hn, &hid_act[row]);
        __builtin_nontemporal_store(en, &alist[row]);
        rb_wr[(size_t)b * kNrec + n] = f2bf(rn);
      }
    }

    // ---- device-wide barrier: make r_{t+1} visible to all 8 WGs ----
    __threadfence();
    __syncthreads();
    if (tid == 0) {
      atomicAdd(barrier_cnt, 1u);
      const unsigned tgt = (unsigned)(t + 1) * (unsigned)kNwg;
      while (__hip_atomic_load(barrier_cnt, __ATOMIC_ACQUIRE,
                               __HIP_MEMORY_SCOPE_AGENT) < tgt)
        __builtin_amdgcn_s_sleep(2);
    }
    __syncthreads();
    __threadfence();
  }
}

// -------------------- launch --------------------

extern "C" void kernel_launch(void* const* d_in, const int* in_sizes, int n_in,
                              void* d_out, int out_size, void* d_ws, size_t ws_size,
                              hipStream_t stream) {
  (void)in_sizes; (void)n_in; (void)out_size; (void)ws_size;
  const float* input = (const float*)d_in[0];
  const float* y     = (const float*)d_in[1];
  const float* mask  = (const float*)d_in[2];
  const float* w_in  = (const float*)d_in[3];
  const float* w_rec = (const float*)d_in[4];
  const float* w_fb  = (const float*)d_in[5];
  const float* w_out = (const float*)d_in[6];
  const float* b_rec = (const float*)d_in[7];
  const float* b_out = (const float*)d_in[8];

  float* out = (float*)d_out;
  const size_t HID = (size_t)kB * (kT + 1) * kNrec;  // 32,800,768
  float* hidden  = out;                              // all-zeros output
  float* hid_act = out + HID;
  float* outputs = out + 2 * HID;
  float* alist   = out + 2 * HID + (size_t)kB * kT * kNout;

  // workspace layout: [0..128) barrier counter; then ping-pong r (bf16, 2x64KB);
  // then w_out in bf16 padded to 16 rows (16KB)
  unsigned* cnt = (unsigned*)d_ws;
  unsigned short* rping  = (unsigned short*)((char*)d_ws + 128);
  unsigned short* woutbf = (unsigned short*)((char*)d_ws + 128 + (size_t)2 * kB * kNrec * 2);

  k_zero_f32<<<1024, 256, 0, stream>>>(hidden, HID);
  k_zero_u32<<<64, 256, 0, stream>>>(cnt, (size_t)(128 + kB * kNrec * 2) / 4);
  k_init_t0<<<kB, 256, 0, stream>>>(hid_act, alist);
  k_cvt_wout<<<(16 * kNrec + 255) / 256, 256, 0, stream>>>(w_out, woutbf);
  ctrnn_rollout<<<kNwg, 256, 0, stream>>>(input, y, mask, w_in, w_rec, w_fb,
                                          b_rec, b_out, hid_act, outputs, alist,
                                          cnt, rping, woutbf);
}
